// MultiHeadAttention_21199958573898
// MI455X (gfx1250) — matrix operand order repaired and checked by reading, not compiled, not run on hardware
//
#include <hip/hip_runtime.h>

// ---------------------------------------------------------------------------
// Transformer-XL relative multi-head attention, fused flash-style, for
// MI455X (gfx1250, wave32, WMMA). All GEMMs use v_wmma_f32_16x16x32_bf16.
// Fragment loads are explicitly double-buffered across loop iterations so the
// scheduler cannot collapse them onto one register set (loads stay in flight
// while the previous tile's WMMAs execute).
// ---------------------------------------------------------------------------

#define B_ 4
#define T_ 2048
#define C_ 512
#define H_ 8
#define D_ 64

typedef __attribute__((ext_vector_type(16))) __bf16 bf16x16;
typedef __attribute__((ext_vector_type(8)))  float  f32x8;

union FragU { bf16x16 v; uint4 q[2]; };
union PackU { __bf16 h[8]; uint4 q; };

// B-fragment (K x 16): lane holds 16 contiguous K values for its column.
static __device__ __forceinline__ bf16x16 load_b_frag(const __bf16* p) {
    FragU u;
    u.q[0] = *(const uint4*)(p);
    u.q[1] = *(const uint4*)(p + 8);
    return u.v;
}

// A-fragment (16 x 32): lane (row = lane&15, g = lane>>4) holds
// K = 16g..16g+15 of its row.
static __device__ __forceinline__ bf16x16 load_a_frag(const __bf16* rowp, int g) {
    FragU u;
    u.q[0] = *(const uint4*)(rowp + 16 * g);
    u.q[1] = *(const uint4*)(rowp + 16 * g + 8);
    return u.v;
}

static __device__ __forceinline__ f32x8 wmma_bf16(bf16x16 a, bf16x16 b, f32x8 c) {
    return __builtin_amdgcn_wmma_f32_16x16x32_bf16(false, a, false, b, (short)0, c,
                                                   false, false);
}

static __device__ __forceinline__ f32x8 zero8() {
    f32x8 z = {0.f, 0.f, 0.f, 0.f, 0.f, 0.f, 0.f, 0.f};
    return z;
}

// ---------------------------------------------------------------------------
// Prep: f32 -> bf16 cast (optionally scaled)
// ---------------------------------------------------------------------------
__global__ void cast_bf16_kernel(const float* __restrict__ s, __bf16* __restrict__ d,
                                 int nmax, float scale) {
    for (int i = blockIdx.x * blockDim.x + threadIdx.x; i < nmax;
         i += gridDim.x * blockDim.x)
        d[i] = (__bf16)(s[i] * scale);
}

// Batched transpose + cast: in [batch][rows][cols] -> out [batch][cols][rows]
__global__ void tcast_bf16_kernel(const float* __restrict__ s, __bf16* __restrict__ d,
                                  int batch, int rows, int cols, float scale) {
    int nmax = batch * rows * cols;
    for (int i = blockIdx.x * blockDim.x + threadIdx.x; i < nmax;
         i += gridDim.x * blockDim.x) {
        int bi  = i / (rows * cols);
        int rem = i - bi * rows * cols;
        int r   = rem / cols;
        int c   = rem - r * cols;
        d[(size_t)bi * rows * cols + (size_t)c * rows + r] = (__bf16)(s[i] * scale);
    }
}

// ---------------------------------------------------------------------------
// QKV projection. One wave computes a 16(t) x 64(d) tile of q, k, v for one
// (b,h). Double-buffered K loop (full unroll: ping/pong indices fold to
// constants). V is written transposed ([D][T]) for P*V B-fragments later.
// ---------------------------------------------------------------------------
__global__ __launch_bounds__(256) void qkv_kernel(
    const __bf16* __restrict__ xb,  const __bf16* __restrict__ wqT,
    const __bf16* __restrict__ wkT, const __bf16* __restrict__ wvT,
    __bf16* __restrict__ q, __bf16* __restrict__ k, __bf16* __restrict__ vT) {
    int lane = threadIdx.x & 31, wid = threadIdx.x >> 5;
    int w     = blockIdx.x * 8 + wid;
    int ttile = w & (T_ / 16 - 1);     // 128 tiles -> 7 bits
    int bh    = w >> 7;
    int b = bh >> 3, h = bh & 7;
    int t0 = ttile * 16;
    int n = lane & 15, g = lane >> 4;

    f32x8 aq[4], ak[4], av[4];
#pragma unroll
    for (int i = 0; i < 4; ++i) { aq[i] = zero8(); ak[i] = zero8(); av[i] = zero8(); }

    const __bf16* xrow = xb + ((size_t)b * T_ + t0 + n) * C_;

    bf16x16 bq[2][4], bk[2][4], bv[2][4], af[2];
    auto ld = [&](int buf, int c0) {
#pragma unroll
        for (int nt = 0; nt < 4; ++nt) {
            const size_t wrow = ((size_t)h * D_ + nt * 16 + n) * C_ + c0 + 16 * g;
            bq[buf][nt] = load_b_frag(wqT + wrow);
            bk[buf][nt] = load_b_frag(wkT + wrow);
            bv[buf][nt] = load_b_frag(wvT + wrow);
        }
        af[buf] = load_a_frag(xrow + c0, g);
    };

    ld(0, 0);
#pragma unroll
    for (int i = 0; i < 16; ++i) {
        const int cur = i & 1, nxt = 1 - cur;
        const int cn  = (i == 15) ? i * 32 : (i + 1) * 32;  // clamp: last reloads self
        ld(nxt, cn);
#pragma unroll
        for (int nt = 0; nt < 4; ++nt) {
            aq[nt] = wmma_bf16(af[cur], bq[cur][nt], aq[nt]);
            ak[nt] = wmma_bf16(af[cur], bk[cur][nt], ak[nt]);
            av[nt] = wmma_bf16(af[cur], bv[cur][nt], av[nt]);
        }
    }

    size_t plane = (size_t)(b * H_ + h);
#pragma unroll
    for (int nt = 0; nt < 4; ++nt) {
        PackU pk;
#pragma unroll
        for (int r = 0; r < 8; ++r) {
            int m = r + 8 * g, t = t0 + m, d = nt * 16 + n;
            q[(plane * T_ + t) * D_ + d] = (__bf16)aq[nt][r];
            k[(plane * T_ + t) * D_ + d] = (__bf16)ak[nt][r];
            pk.h[r] = (__bf16)av[nt][r];
        }
        // v^T store: 8 consecutive t per lane -> one b128 store
        *(uint4*)(vT + (plane * D_ + nt * 16 + n) * T_ + t0 + 8 * g) = pk.q;
    }
}

// ---------------------------------------------------------------------------
// Fused flash attention with Transformer-XL relative positions.
// One wave owns one 16-query tile; loops over 32-wide key tiles.
//   S  = q * k^T                     (2 N-halves x 2 K-chunks WMMA)
//   R  = q * rel[jbase..jbase+47]^T  (3 N-tiles x 2 K-chunks WMMA)
//   BD[t,s] = R[t, (s-t)+15]  via per-wave LDS skewed read (rel_shift)
//   online softmax (shfl_xor width-16 row reductions), O += P*V (4 WMMA)
// k/rel fragments double-buffered across key tiles (manual 2x rotation with
// distinct array sets so no dynamic indexing / scratch spills).
// ---------------------------------------------------------------------------
__global__ __launch_bounds__(256) void attn_kernel(
    const __bf16* __restrict__ q, const __bf16* __restrict__ k,
    const __bf16* __restrict__ vT, const __bf16* __restrict__ relb,
    __bf16* __restrict__ ao) {
    __shared__ float  Rs[8][16][49];   // per-wave banded rel scores
    __shared__ __bf16 Ps[8][16][32];   // per-wave P tile (A-layout staging)

    int lane = threadIdx.x & 31, wid = threadIdx.x >> 5;
    int w     = blockIdx.x * 8 + wid;
    int ttile = w & 127;
    int bh    = w >> 7;
    int b = bh >> 3, h = bh & 7;
    int t0 = ttile * 16;
    int n = lane & 15, g = lane >> 4;

    const __bf16* qp = q    + (size_t)(b * H_ + h) * T_ * D_;
    const __bf16* kp = k    + (size_t)(b * H_ + h) * T_ * D_;
    const __bf16* vp = vT   + (size_t)(b * H_ + h) * D_ * T_;
    const __bf16* rp = relb + (size_t)h * T_ * D_;

    bf16x16 aqf[2];
    {
        const __bf16* qrow = qp + (size_t)(t0 + n) * D_;
        aqf[0] = load_a_frag(qrow, g);
        aqf[1] = load_a_frag(qrow + 32, g);
    }

    f32x8 accO[4];
#pragma unroll
    for (int i = 0; i < 4; ++i) accO[i] = zero8();
    float mx[8], ls[8];
#pragma unroll
    for (int r = 0; r < 8; ++r) { mx[r] = -3.0e38f; ls[r] = 0.f; }
    const float L2E = 1.4426950408889634f;
    const int send = t0 + 16;

    // next-tile fragment loader (10 frags: 2 K-chunks x (2 k + 3 rel))
    auto ldtile = [&](bf16x16 (&bkf)[2][2], bf16x16 (&brf)[2][3], int s0) {
        int jbase = (T_ - 16) + s0 - t0;
        const __bf16* krow[2];
#pragma unroll
        for (int h2 = 0; h2 < 2; ++h2) {
            int s = s0 + 16 * h2 + n;
            if (s > T_ - 1) s = T_ - 1;  // masked anyway; keep loads in-bounds
            krow[h2] = kp + (size_t)s * D_ + 16 * g;
        }
        const __bf16* rrow[3];
#pragma unroll
        for (int jt = 0; jt < 3; ++jt) {
            int j = jbase + 16 * jt + n;
            j = j < 0 ? 0 : (j > T_ - 1 ? T_ - 1 : j);
            rrow[jt] = rp + (size_t)j * D_ + 16 * g;
        }
#pragma unroll
        for (int ci = 0; ci < 2; ++ci) {
#pragma unroll
            for (int h2 = 0; h2 < 2; ++h2) bkf[ci][h2] = load_b_frag(krow[h2] + 32 * ci);
#pragma unroll
            for (int jt = 0; jt < 3; ++jt) brf[ci][jt] = load_b_frag(rrow[jt] + 32 * ci);
        }
    };

    auto compute = [&](bf16x16 (&bkf)[2][2], bf16x16 (&brf)[2][3], int s0) {
        // V fragments issued first, consumed only after softmax -> latency hidden
        bf16x16 bv[4];
#pragma unroll
        for (int nt = 0; nt < 4; ++nt)
            bv[nt] = load_b_frag(vp + (size_t)(nt * 16 + n) * T_ + s0 + 16 * g);

        f32x8 sA[2]; sA[0] = zero8(); sA[1] = zero8();
        f32x8 rA[3]; rA[0] = zero8(); rA[1] = zero8(); rA[2] = zero8();
#pragma unroll
        for (int ci = 0; ci < 2; ++ci) {
#pragma unroll
            for (int h2 = 0; h2 < 2; ++h2)
                sA[h2] = wmma_bf16(aqf[ci], bkf[ci][h2], sA[h2]);
#pragma unroll
            for (int jt = 0; jt < 3; ++jt)
                rA[jt] = wmma_bf16(aqf[ci], brf[ci][jt], rA[jt]);
        }

        // stage banded rel scores (per-wave region; same-wave DS ops in order)
#pragma unroll
        for (int jt = 0; jt < 3; ++jt)
#pragma unroll
            for (int r = 0; r < 8; ++r)
                Rs[wid][r + 8 * g][jt * 16 + n] = rA[jt][r];

        // combine + rel-shift + mask
        float sv[2][8];
#pragma unroll
        for (int h2 = 0; h2 < 2; ++h2)
#pragma unroll
            for (int r = 0; r < 8; ++r) {
                int m = r + 8 * g;
                int s = s0 + 16 * h2 + n;
                float val = sA[h2][r] + Rs[wid][m][16 * h2 + n - m + 15];
                bool ok = (s <= t0 + m) || (s < 6);
                if (!ok || s >= T_) val = -3.0e38f;
                sv[h2][r] = val;
            }

        // online softmax, row stats live across 16-lane half-wave groups
#pragma unroll
        for (int r = 0; r < 8; ++r) {
            float rmax = fmaxf(sv[0][r], sv[1][r]);
#pragma unroll
            for (int off = 1; off < 16; off <<= 1)
                rmax = fmaxf(rmax, __shfl_xor(rmax, off, 16));
            float mnew  = fmaxf(mx[r], rmax);
            float alpha = exp2f((mx[r] - mnew) * L2E);
            mx[r] = mnew;
            float p0 = exp2f((sv[0][r] - mnew) * L2E);
            float p1 = exp2f((sv[1][r] - mnew) * L2E);
            float psum = p0 + p1;
#pragma unroll
            for (int off = 1; off < 16; off <<= 1)
                psum += __shfl_xor(psum, off, 16);
            ls[r] = ls[r] * alpha + psum;
#pragma unroll
            for (int nt = 0; nt < 4; ++nt) accO[nt][r] *= alpha;
            int m = r + 8 * g;
            Ps[wid][m][n]      = (__bf16)p0;
            Ps[wid][m][16 + n] = (__bf16)p1;
        }

        // O += P * V   (V^T rows give contiguous B-fragments)
        bf16x16 pf = load_a_frag(&Ps[wid][n][0], g);
#pragma unroll
        for (int nt = 0; nt < 4; ++nt)
            accO[nt] = wmma_bf16(pf, bv[nt], accO[nt]);
    };

    // two distinct fragment sets; manual 2x rotation (no dynamic buffer index)
    bf16x16 bkA[2][2], brA3[2][3], bkB[2][2], brB3[2][3];
    ldtile(bkA, brA3, 0);
    int s0 = 0;
    for (;;) {
        int s0n = (s0 + 32 < send) ? s0 + 32 : s0;   // last iter: clamped reload
        ldtile(bkB, brB3, s0n);
        __builtin_prefetch(kp + (size_t)((s0 + 64 < send) ? s0 + 64 : s0) * D_, 0, 1);
        compute(bkA, brA3, s0);
        s0 += 32;
        if (s0 >= send) break;

        s0n = (s0 + 32 < send) ? s0 + 32 : s0;
        ldtile(bkA, brA3, s0n);
        compute(bkB, brB3, s0);
        s0 += 32;
        if (s0 >= send) break;
    }

#pragma unroll
    for (int r = 0; r < 8; ++r) {
        float inv = 1.f / ls[r];
        int m = r + 8 * g, t = t0 + m;
#pragma unroll
        for (int nt = 0; nt < 4; ++nt)
            ao[(size_t)(b * T_ + t) * C_ + h * D_ + nt * 16 + n] =
                (__bf16)(accO[nt][r] * inv);
    }
}

// ---------------------------------------------------------------------------
// Output projection: [B*T, C] x [C, C] + bias -> f32. Double-buffered K loop.
// ---------------------------------------------------------------------------
__global__ __launch_bounds__(256) void proj_kernel(
    const __bf16* __restrict__ ao, const __bf16* __restrict__ wpT,
    const float* __restrict__ bp, float* __restrict__ out) {
    int lane = threadIdx.x & 31, wid = threadIdx.x >> 5;
    int n = lane & 15, g = lane >> 4;
    size_t row0 = (size_t)blockIdx.x * 16;
    int    col0 = wid * 64;

    f32x8 acc[4];
#pragma unroll
    for (int i = 0; i < 4; ++i) acc[i] = zero8();

    const __bf16* arow = ao + (row0 + n) * C_;

    bf16x16 bw[2][4], af[2];
    auto ld = [&](int buf, int c0) {
#pragma unroll
        for (int nt = 0; nt < 4; ++nt)
            bw[buf][nt] =
                load_b_frag(wpT + (size_t)(col0 + nt * 16 + n) * C_ + c0 + 16 * g);
        af[buf] = load_a_frag(arow + c0, g);
    };

    ld(0, 0);
#pragma unroll
    for (int i = 0; i < 16; ++i) {
        const int cur = i & 1, nxt = 1 - cur;
        const int cn  = (i == 15) ? i * 32 : (i + 1) * 32;
        ld(nxt, cn);
#pragma unroll
        for (int nt = 0; nt < 4; ++nt)
            acc[nt] = wmma_bf16(af[cur], bw[cur][nt], acc[nt]);
    }

#pragma unroll
    for (int nt = 0; nt < 4; ++nt) {
        int col = col0 + nt * 16 + n;
        float bias = bp[col];
#pragma unroll
        for (int r = 0; r < 8; ++r) {
            int m = r + 8 * g;
            out[(row0 + m) * C_ + col] = acc[nt][r] + bias;
        }
    }
}

// ---------------------------------------------------------------------------
extern "C" void kernel_launch(void* const* d_in, const int* in_sizes, int n_in,
                              void* d_out, int out_size, void* d_ws, size_t ws_size,
                              hipStream_t stream) {
    const float* x   = (const float*)d_in[0];
    const float* Wq  = (const float*)d_in[1];
    const float* Wk  = (const float*)d_in[2];
    const float* Wv  = (const float*)d_in[3];
    const float* rel = (const float*)d_in[4];
    const float* Wp  = (const float*)d_in[5];
    const float* bp  = (const float*)d_in[6];
    float* out = (float*)d_out;

    char* ws = (char*)d_ws;
    size_t off = 0;
    auto carve = [&](size_t bytes) -> char* {
        char* p = ws + off;
        off += (bytes + 255) & ~(size_t)255;
        return p;
    };
    __bf16* xb   = (__bf16*)carve((size_t)B_ * T_ * C_ * 2);
    __bf16* wqT  = (__bf16*)carve((size_t)H_ * D_ * C_ * 2);
    __bf16* wkT  = (__bf16*)carve((size_t)H_ * D_ * C_ * 2);
    __bf16* wvT  = (__bf16*)carve((size_t)H_ * D_ * C_ * 2);
    __bf16* wpT  = (__bf16*)carve((size_t)C_ * C_ * 2);
    __bf16* relb = (__bf16*)carve((size_t)H_ * T_ * D_ * 2);
    __bf16* qb   = (__bf16*)carve((size_t)B_ * H_ * T_ * D_ * 2);
    __bf16* kb   = (__bf16*)carve((size_t)B_ * H_ * T_ * D_ * 2);
    __bf16* vtb  = (__bf16*)carve((size_t)B_ * H_ * D_ * T_ * 2 + 256); // OOB pad
    __bf16* aob  = (__bf16*)carve((size_t)B_ * T_ * C_ * 2);
    (void)ws_size; (void)in_sizes; (void)n_in; (void)out_size;

    const float qscale = 0.044194173824159216f;  // 512^-0.5 folded into Wq

    cast_bf16_kernel<<<2048, 256, 0, stream>>>(x, xb, B_ * T_ * C_, 1.f);
    cast_bf16_kernel<<<1024, 256, 0, stream>>>(rel, relb, H_ * T_ * D_, 1.f);
    tcast_bf16_kernel<<<512, 256, 0, stream>>>(Wq, wqT, H_, C_, D_, qscale);
    tcast_bf16_kernel<<<512, 256, 0, stream>>>(Wk, wkT, H_, C_, D_, 1.f);
    tcast_bf16_kernel<<<512, 256, 0, stream>>>(Wv, wvT, H_, C_, D_, 1.f);
    tcast_bf16_kernel<<<512, 256, 0, stream>>>(Wp, wpT, 1, C_, C_, 1.f);

    qkv_kernel<<<B_ * H_ * (T_ / 16) / 8, 256, 0, stream>>>(xb, wqT, wkT, wvT,
                                                            qb, kb, vtb);
    attn_kernel<<<B_ * H_ * (T_ / 16) / 8, 256, 0, stream>>>(qb, kb, vtb, relb, aob);
    proj_kernel<<<B_ * T_ / 16, 256, 0, stream>>>(aob, wpT, bp, out);
}
